// AttentionWithReturn_1846835938112
// MI455X (gfx1250) — matrix operand order repaired and checked by reading, not compiled, not run on hardware
//
#include <hip/hip_runtime.h>
#include <hip/hip_bf16.h>

// MI455X (gfx1250) fused multi-head attention:
//   B=8, N=1024, C=768, H=12, D=64
// Pipeline: f32->f16 convert -> QKV WMMA GEMM (2x2 register-blocked) ->
//           flash attention (WMMA) -> proj WMMA GEMM (2x2 register-blocked).

typedef __attribute__((ext_vector_type(16))) _Float16 v16h;
typedef __attribute__((ext_vector_type(8)))  _Float16 v8h;
typedef __attribute__((ext_vector_type(8)))  float    v8f;

#define BB 8
#define NN 1024
#define CC 768
#define HH 12
#define DD 64

static __device__ inline v8f wmma_f16(v16h a, v16h b, v8f c) {
    return __builtin_amdgcn_wmma_f32_16x16x32_f16(false, a, false, b, (short)0, c,
                                                  false, false);
}

// ---------------------------------------------------------------- convert
__global__ void f32_to_f16_kernel(const float* __restrict__ src,
                                  _Float16* __restrict__ dst, int n) {
    int i = blockIdx.x * blockDim.x + threadIdx.x;
    int stride = gridDim.x * blockDim.x;
    for (; i < n; i += stride) dst[i] = (_Float16)src[i];
}

// ---------------------------------------------------------------- QKV GEMM
// out[m, c] = sum_k xh[m,k] * wqh[c,k]   (M=8192, K=768, Nout=2304)
// Block: 256 thr = 8 waves in 2x4; wave computes 32x32 (4 C frags).
// Macro tile 64(M) x 128(N). Scatter into Qh/Kh/Vh [B,H,N,D] f16 (scale in Q).
__global__ __launch_bounds__(256) void qkv_gemm_kernel(
    const _Float16* __restrict__ xh, const _Float16* __restrict__ wqh,
    _Float16* __restrict__ Qh, _Float16* __restrict__ Kh,
    _Float16* __restrict__ Vh) {
    const int M0 = blockIdx.x * 64;
    const int N0 = blockIdx.y * 128;

    __shared__ _Float16 As[64][80];    // A tile: [m][k], pitch 160B
    __shared__ _Float16 Bs[128][80];   // Bt tile: [n(out col)][k]

    const int tid = threadIdx.x;
    const int lane = tid & 31;
    const int wave = tid >> 5;
    const int hf = lane >> 4;
    const int ln = lane & 15;
    const int b8 = hf * 8;
    const int b16 = hf * 16;
    const int wr = wave >> 2;   // 0..1 -> M offset wr*32
    const int wc = wave & 3;    // 0..3 -> N offset wc*32

    v8f acc00 = {0.f,0.f,0.f,0.f,0.f,0.f,0.f,0.f};
    v8f acc01 = acc00, acc10 = acc00, acc11 = acc00;

    for (int k0 = 0; k0 < CC; k0 += 64) {
        __syncthreads();
        {   // stage A: 64x64 (16 f16 per thread)
            int r = tid >> 2, c = (tid & 3) * 16;
            const _Float16* src = &xh[(size_t)(M0 + r) * CC + k0 + c];
            *(v8h*)&As[r][c]     = *(const v8h*)(src);
            *(v8h*)&As[r][c + 8] = *(const v8h*)(src + 8);
            if (k0 + 64 < CC) __builtin_prefetch(src + 64, 0, 3);
        }
        {   // stage Bt: 128x64 (32 f16 per thread); weight row-major == Bt
            int r = tid >> 1, c = (tid & 1) * 32;
            const _Float16* src = &wqh[(size_t)(N0 + r) * CC + k0 + c];
            *(v8h*)&Bs[r][c]      = *(const v8h*)(src);
            *(v8h*)&Bs[r][c + 8]  = *(const v8h*)(src + 8);
            *(v8h*)&Bs[r][c + 16] = *(const v8h*)(src + 16);
            *(v8h*)&Bs[r][c + 24] = *(const v8h*)(src + 24);
            if (k0 + 64 < CC) __builtin_prefetch(src + 64, 0, 3);
        }
        __syncthreads();
#pragma unroll
        for (int kk = 0; kk < 64; kk += 32) {
            v16h a0, a1, b0, b1;
            ((v8h*)&a0)[0] = *(v8h*)&As[wr * 32 + ln][kk + b16];
            ((v8h*)&a0)[1] = *(v8h*)&As[wr * 32 + ln][kk + b16 + 8];
            ((v8h*)&a1)[0] = *(v8h*)&As[wr * 32 + 16 + ln][kk + b16];
            ((v8h*)&a1)[1] = *(v8h*)&As[wr * 32 + 16 + ln][kk + b16 + 8];
            ((v8h*)&b0)[0] = *(v8h*)&Bs[wc * 32 + ln][kk + b16];
            ((v8h*)&b0)[1] = *(v8h*)&Bs[wc * 32 + ln][kk + b16 + 8];
            ((v8h*)&b1)[0] = *(v8h*)&Bs[wc * 32 + 16 + ln][kk + b16];
            ((v8h*)&b1)[1] = *(v8h*)&Bs[wc * 32 + 16 + ln][kk + b16 + 8];
            acc00 = wmma_f16(a0, b0, acc00);
            acc01 = wmma_f16(a0, b1, acc01);
            acc10 = wmma_f16(a1, b0, acc10);
            acc11 = wmma_f16(a1, b1, acc11);
        }
    }

    // epilogue: C frag element r -> (row = r + 8*hf, col = ln) within 16x16 tile
    v8f* accs[2][2] = {{&acc00, &acc01}, {&acc10, &acc11}};
#pragma unroll
    for (int i = 0; i < 2; ++i) {
#pragma unroll
        for (int j = 0; j < 2; ++j) {
            const v8f& a = *accs[i][j];
            const int col = N0 + wc * 32 + j * 16 + ln;
            const int which = col / CC;     // 0=Q 1=K 2=V
            const int wi = col % CC;
            const int h = wi >> 6, d = wi & 63;
            _Float16* dst = (which == 0) ? Qh : ((which == 1) ? Kh : Vh);
            const float scl = (which == 0) ? 0.125f : 1.0f;
#pragma unroll
            for (int r = 0; r < 8; ++r) {
                int row = M0 + wr * 32 + i * 16 + r + 8 * hf;
                int bb = row >> 10, nq = row & 1023;
                size_t idx = (((size_t)(bb * HH + h)) * NN + nq) * DD + d;
                dst[idx] = (_Float16)(a[r] * scl);
            }
        }
    }
}

// ---------------------------------------------------------------- attention
// One block = (b, h, 128 query rows); one wave = 16 query rows.
__global__ __launch_bounds__(256) void attn_kernel(
    const _Float16* __restrict__ Qh, const _Float16* __restrict__ Kh,
    const _Float16* __restrict__ Vh, _Float16* __restrict__ Oh) {
    const int blk = blockIdx.x;
    const int qb = blk & 7;            // N/128 = 8 query blocks
    const int bh = blk >> 3;           // 0..95
    const int h = bh % HH, b = bh / HH;
    const size_t base = (size_t)(b * HH + h) * NN * DD;

    const int tid = threadIdx.x;
    const int lane = tid & 31;
    const int wave = tid >> 5;
    const int hf = lane >> 4;
    const int ln = lane & 15;
    const int b8 = hf * 8;
    const int b16 = hf * 16;
    const int q0 = qb * 128 + wave * 16;

    __shared__ _Float16 Ks[32][80];        // K tile row-major [key][d]
    __shared__ _Float16 Vt[64][40];        // V tile transposed [d][key]
    __shared__ _Float16 Ps[8][16][40];     // per-wave P tile [m][key]

    // Q fragments (A layout) straight from global: row = q0+ln
    v16h aq0, aq1;
    {
        const _Float16* qrow = Qh + base + (size_t)(q0 + ln) * DD;
        ((v8h*)&aq0)[0] = *(const v8h*)(qrow + b16);
        ((v8h*)&aq0)[1] = *(const v8h*)(qrow + b16 + 8);
        ((v8h*)&aq1)[0] = *(const v8h*)(qrow + 32 + b16);
        ((v8h*)&aq1)[1] = *(const v8h*)(qrow + 32 + b16 + 8);
    }

    v8f o0 = {0.f,0.f,0.f,0.f,0.f,0.f,0.f,0.f};
    v8f o1 = o0, o2 = o0, o3 = o0;
    float ms[8], ls[8];
#pragma unroll
    for (int r = 0; r < 8; ++r) { ms[r] = -1e30f; ls[r] = 0.f; }

    for (int k0 = 0; k0 < NN; k0 += 32) {
        {   // stage K (row-major, directly a Bt tile) and V transposed
            int i = tid >> 3, c = (tid & 7) * 8;
            *(v8h*)&Ks[i][c] = *(const v8h*)(Kh + base + (size_t)(k0 + i) * DD + c);
            v8h vv = *(const v8h*)(Vh + base + (size_t)(k0 + i) * DD + c);
#pragma unroll
            for (int u = 0; u < 8; ++u) Vt[c + u][i] = vv[u];
        }
        __syncthreads();

        // S = Q * K^T : two 16x16 tiles (keys 0-15, 16-31), contraction d=64
        v8f s0 = {0.f,0.f,0.f,0.f,0.f,0.f,0.f,0.f};
        v8f s1 = s0;
        {
            v16h bk;
            ((v8h*)&bk)[0] = *(v8h*)&Ks[ln][b16];
            ((v8h*)&bk)[1] = *(v8h*)&Ks[ln][b16 + 8];
            s0 = wmma_f16(aq0, bk, s0);
            ((v8h*)&bk)[0] = *(v8h*)&Ks[ln][32 + b16];
            ((v8h*)&bk)[1] = *(v8h*)&Ks[ln][32 + b16 + 8];
            s0 = wmma_f16(aq1, bk, s0);
            ((v8h*)&bk)[0] = *(v8h*)&Ks[16 + ln][b16];
            ((v8h*)&bk)[1] = *(v8h*)&Ks[16 + ln][b16 + 8];
            s1 = wmma_f16(aq0, bk, s1);
            ((v8h*)&bk)[0] = *(v8h*)&Ks[16 + ln][32 + b16];
            ((v8h*)&bk)[1] = *(v8h*)&Ks[16 + ln][32 + b16 + 8];
            s1 = wmma_f16(aq1, bk, s1);
        }

        // online softmax per row (row = r + 8*hf; 16 lanes of the half hold its cols)
#pragma unroll
        for (int r = 0; r < 8; ++r) {
            float x0 = s0[r], x1 = s1[r];
            float rm = fmaxf(x0, x1);
            rm = fmaxf(rm, __shfl_xor(rm, 1, 32));
            rm = fmaxf(rm, __shfl_xor(rm, 2, 32));
            rm = fmaxf(rm, __shfl_xor(rm, 4, 32));
            rm = fmaxf(rm, __shfl_xor(rm, 8, 32));
            float mnew = fmaxf(ms[r], rm);
            float p0 = __expf(x0 - mnew);
            float p1 = __expf(x1 - mnew);
            float rs = p0 + p1;
            rs += __shfl_xor(rs, 1, 32);
            rs += __shfl_xor(rs, 2, 32);
            rs += __shfl_xor(rs, 4, 32);
            rs += __shfl_xor(rs, 8, 32);
            float corr = __expf(ms[r] - mnew);
            ls[r] = ls[r] * corr + rs;
            ms[r] = mnew;
            o0[r] *= corr; o1[r] *= corr; o2[r] *= corr; o3[r] *= corr;
            int m = r + 8 * hf;
            Ps[wave][m][ln]      = (_Float16)p0;
            Ps[wave][m][16 + ln] = (_Float16)p1;
        }
        // intra-wave DS RAW: all lanes' P stores must land before A-frag reload
        asm volatile("s_wait_dscnt 0" ::: "memory");

        // O += P(16x32) * V(32x64)
        v16h pa;
        ((v8h*)&pa)[0] = *(v8h*)&Ps[wave][ln][b16];
        ((v8h*)&pa)[1] = *(v8h*)&Ps[wave][ln][b16 + 8];
        {
            v16h bv;
            ((v8h*)&bv)[0] = *(v8h*)&Vt[0 * 16 + ln][b16];
            ((v8h*)&bv)[1] = *(v8h*)&Vt[0 * 16 + ln][b16 + 8];
            o0 = wmma_f16(pa, bv, o0);
            ((v8h*)&bv)[0] = *(v8h*)&Vt[1 * 16 + ln][b16];
            ((v8h*)&bv)[1] = *(v8h*)&Vt[1 * 16 + ln][b16 + 8];
            o1 = wmma_f16(pa, bv, o1);
            ((v8h*)&bv)[0] = *(v8h*)&Vt[2 * 16 + ln][b16];
            ((v8h*)&bv)[1] = *(v8h*)&Vt[2 * 16 + ln][b16 + 8];
            o2 = wmma_f16(pa, bv, o2);
            ((v8h*)&bv)[0] = *(v8h*)&Vt[3 * 16 + ln][b16];
            ((v8h*)&bv)[1] = *(v8h*)&Vt[3 * 16 + ln][b16 + 8];
            o3 = wmma_f16(pa, bv, o3);
        }
        __syncthreads();
    }

    // finalize: divide by l, write Oh [B,N,C] f16 (token-major for the proj GEMM)
#pragma unroll
    for (int r = 0; r < 8; ++r) {
        float inv = 1.0f / ls[r];
        int row = q0 + r + 8 * hf;
        size_t orow = ((size_t)(b * NN + row)) * CC + h * DD;
        Oh[orow + 0 * 16 + ln] = (_Float16)(o0[r] * inv);
        Oh[orow + 1 * 16 + ln] = (_Float16)(o1[r] * inv);
        Oh[orow + 2 * 16 + ln] = (_Float16)(o2[r] * inv);
        Oh[orow + 3 * 16 + ln] = (_Float16)(o3[r] * inv);
    }
}

// ---------------------------------------------------------------- out proj
// out[m, c] = sum_k Oh[m,k] * wph[c,k] + bias[c]   (M=8192, K=768, N=768), f32 out
// Same 2x2 register blocking: macro tile 64 x 128.
__global__ __launch_bounds__(256) void proj_gemm_kernel(
    const _Float16* __restrict__ Oh, const _Float16* __restrict__ wph,
    const float* __restrict__ bias, float* __restrict__ out) {
    const int M0 = blockIdx.x * 64;
    const int N0 = blockIdx.y * 128;

    __shared__ _Float16 As[64][80];
    __shared__ _Float16 Bs[128][80];

    const int tid = threadIdx.x;
    const int lane = tid & 31;
    const int wave = tid >> 5;
    const int hf = lane >> 4;
    const int ln = lane & 15;
    const int b8 = hf * 8;
    const int b16 = hf * 16;
    const int wr = wave >> 2;
    const int wc = wave & 3;

    v8f acc00 = {0.f,0.f,0.f,0.f,0.f,0.f,0.f,0.f};
    v8f acc01 = acc00, acc10 = acc00, acc11 = acc00;

    for (int k0 = 0; k0 < CC; k0 += 64) {
        __syncthreads();
        {
            int r = tid >> 2, c = (tid & 3) * 16;
            const _Float16* src = &Oh[(size_t)(M0 + r) * CC + k0 + c];
            *(v8h*)&As[r][c]     = *(const v8h*)(src);
            *(v8h*)&As[r][c + 8] = *(const v8h*)(src + 8);
            if (k0 + 64 < CC) __builtin_prefetch(src + 64, 0, 3);
        }
        {
            int r = tid >> 1, c = (tid & 1) * 32;
            const _Float16* src = &wph[(size_t)(N0 + r) * CC + k0 + c];
            *(v8h*)&Bs[r][c]      = *(const v8h*)(src);
            *(v8h*)&Bs[r][c + 8]  = *(const v8h*)(src + 8);
            *(v8h*)&Bs[r][c + 16] = *(const v8h*)(src + 16);
            *(v8h*)&Bs[r][c + 24] = *(const v8h*)(src + 24);
            if (k0 + 64 < CC) __builtin_prefetch(src + 64, 0, 3);
        }
        __syncthreads();
#pragma unroll
        for (int kk = 0; kk < 64; kk += 32) {
            v16h a0, a1, b0, b1;
            ((v8h*)&a0)[0] = *(v8h*)&As[wr * 32 + ln][kk + b16];
            ((v8h*)&a0)[1] = *(v8h*)&As[wr * 32 + ln][kk + b16 + 8];
            ((v8h*)&a1)[0] = *(v8h*)&As[wr * 32 + 16 + ln][kk + b16];
            ((v8h*)&a1)[1] = *(v8h*)&As[wr * 32 + 16 + ln][kk + b16 + 8];
            ((v8h*)&b0)[0] = *(v8h*)&Bs[wc * 32 + ln][kk + b16];
            ((v8h*)&b0)[1] = *(v8h*)&Bs[wc * 32 + ln][kk + b16 + 8];
            ((v8h*)&b1)[0] = *(v8h*)&Bs[wc * 32 + 16 + ln][kk + b16];
            ((v8h*)&b1)[1] = *(v8h*)&Bs[wc * 32 + 16 + ln][kk + b16 + 8];
            acc00 = wmma_f16(a0, b0, acc00);
            acc01 = wmma_f16(a0, b1, acc01);
            acc10 = wmma_f16(a1, b0, acc10);
            acc11 = wmma_f16(a1, b1, acc11);
        }
    }

    v8f* accs[2][2] = {{&acc00, &acc01}, {&acc10, &acc11}};
#pragma unroll
    for (int i = 0; i < 2; ++i) {
#pragma unroll
        for (int j = 0; j < 2; ++j) {
            const v8f& a = *accs[i][j];
            const int col = N0 + wc * 32 + j * 16 + ln;
            const float bv = bias[col];
#pragma unroll
            for (int r = 0; r < 8; ++r) {
                int row = M0 + wr * 32 + i * 16 + r + 8 * hf;
                out[(size_t)row * CC + col] = a[r] + bv;
            }
        }
    }
}

// ---------------------------------------------------------------- launch
extern "C" void kernel_launch(void* const* d_in, const int* in_sizes, int n_in,
                              void* d_out, int out_size, void* d_ws, size_t ws_size,
                              hipStream_t stream) {
    const float* x      = (const float*)d_in[0];
    const float* w_qkv  = (const float*)d_in[1];
    const float* w_proj = (const float*)d_in[2];
    const float* b_proj = (const float*)d_in[3];
    float* out = (float*)d_out;

    const size_t nX  = (size_t)BB * NN * CC;       // 6,291,456
    const size_t nWq = (size_t)3 * CC * CC;        // 1,769,472
    const size_t nWp = (size_t)CC * CC;            //   589,824

    char* ws = (char*)d_ws;
    size_t off = 0;
    auto alloc = [&](size_t bytes) -> char* {
        char* p = ws + off;
        off += (bytes + 255) & ~(size_t)255;
        return p;
    };
    _Float16* xh  = (_Float16*)alloc(nX * 2);
    _Float16* wqh = (_Float16*)alloc(nWq * 2);
    _Float16* wph = (_Float16*)alloc(nWp * 2);
    _Float16* Qh  = (_Float16*)alloc(nX * 2);
    _Float16* Kh  = (_Float16*)alloc(nX * 2);
    _Float16* Vh  = (_Float16*)alloc(nX * 2);
    _Float16* Oh  = (_Float16*)alloc(nX * 2);

    f32_to_f16_kernel<<<2048, 256, 0, stream>>>(x, xh, (int)nX);
    f32_to_f16_kernel<<<512, 256, 0, stream>>>(w_qkv, wqh, (int)nWq);
    f32_to_f16_kernel<<<256, 256, 0, stream>>>(w_proj, wph, (int)nWp);

    qkv_gemm_kernel<<<dim3(128, 18), 256, 0, stream>>>(xh, wqh, Qh, Kh, Vh);
    attn_kernel<<<768, 256, 0, stream>>>(Qh, Kh, Vh, Oh);
    proj_gemm_kernel<<<dim3(128, 6), 256, 0, stream>>>(Oh, wph, b_proj, out);
}
